// Net_25305947308819
// MI455X (gfx1250) — matrix operand order repaired and checked by reading, not compiled, not run on hardware
//
#include <hip/hip_runtime.h>

typedef __attribute__((ext_vector_type(16))) __bf16 v16bf;
typedef __attribute__((ext_vector_type(8)))  float  v8f;

union BF16x16 { unsigned short h[16]; v16bf v; };

__device__ __forceinline__ unsigned short f2bf(float f) {
  union { float f; unsigned u; } x; x.f = f;
  unsigned r = x.u + 0x7FFFu + ((x.u >> 16) & 1u);  // round-to-nearest-even
  return (unsigned short)(r >> 16);
}
__device__ __forceinline__ float leaky(float f) { return f >= 0.f ? f : 0.01f * f; }
__device__ __forceinline__ int imin(int a, int b) { return a < b ? a : b; }

__device__ __forceinline__ v8f vbroad(float f) {
  v8f r;
#pragma unroll
  for (int v = 0; v < 8; ++v) r[v] = f;
  return r;
}

// B fragment: pre-swizzled [kt][lane][16], one aligned 32B read per lane.
__device__ __forceinline__ v8f wmma_bf16(const BF16x16& a, const unsigned short* bfr, v8f acc) {
  v16bf bv = *(const v16bf*)bfr;
  return __builtin_amdgcn_wmma_f32_16x16x32_bf16(false, a.v, false, bv, (short)0, acc,
                                                 false, false);
}
__device__ __forceinline__ void ld8(BF16x16& a, int lo, const unsigned short* p) {
  union { unsigned short h[8]; uint4 u; } c; c.u = *(const uint4*)p;
#pragma unroll
  for (int i = 0; i < 8; ++i) a.h[lo + i] = c.h[i];
}
__device__ __forceinline__ void ld4(BF16x16& a, int lo, const unsigned short* p) {
  union { unsigned short h[4]; uint2 u; } c; c.u = *(const uint2*)p;
#pragma unroll
  for (int i = 0; i < 4; ++i) a.h[lo + i] = c.h[i];
}

// Padded-tap position offsets (fold to constants under full unroll).
__device__ __forceinline__ int dpos2(int t) { return t < 25 ? (t / 5) * 26 + (t % 5) : 0; }
__device__ __forceinline__ int dpos3(int t) { return t < 9 ? (t / 3) * 11 + (t % 3) : 0; }
__device__ __forceinline__ int dpos4(int t) { return t < 9 ? (t / 3) * 9  + (t % 3) : 0; }

// ---------------------------------------------------------------------------
// LDS pool (ushort units, offsets 32B-aligned).
//  region A (0..13312): conv1 out a1p [16][26][32]; after gaze26 reused for:
//    A2P @0 [16][11][16], K11F @2816, A2T @3328 [121][16], W3F @5264,
//    A3T @7824 [81][16], W4F @9120, A4(f32) @11680
//  region B (13312..32352): conv1 phase W1F @13312, SIN @16896 (46 rows max);
//    after conv1: A1T @13312 [676][16], K26F0 @24128, K26F1 @24640;
//    after gaze26: W2F @24128 (13*512)
//  trash slots (branchless guarded stores): ushort @32352 + lane, f32 @32384*2B + lane
// ---------------------------------------------------------------------------
#define P_A1P    0
#define P_A2P    0
#define P_K11F   2816
#define P_A2T    3328
#define P_W3F    5264
#define P_A3T    7824
#define P_W4F    9120
#define P_A4     11680
#define P_W1F    13312
#define P_SIN    16896
#define P_A1T    13312
#define P_K26F0  24128
#define P_K26F1  24640
#define P_W2F    24128
#define P_TRASH  32352
#define P_TRASHF 32384
#define P_TOT    32448

__global__ __launch_bounds__(256)
void conv_frames(const float* __restrict__ traj_i, const float* __restrict__ traj_j,
                 const float* __restrict__ g26i, const float* __restrict__ g11i,
                 const float* __restrict__ g26j, const float* __restrict__ g11j,
                 const float* __restrict__ w1, const float* __restrict__ b1,
                 const float* __restrict__ w2, const float* __restrict__ b2,
                 const float* __restrict__ w3, const float* __restrict__ b3,
                 const float* __restrict__ w4, const float* __restrict__ b4,
                 unsigned short* __restrict__ ws_xf, float* __restrict__ ws_cm) {
  __shared__ __align__(32) unsigned short s_pool[P_TOT];
  unsigned short* const s_a1p = s_pool + P_A1P;
  unsigned short* const s_a1t = s_pool + P_A1T;
  unsigned short* const s_a2p = s_pool + P_A2P;
  unsigned short* const s_a2t = s_pool + P_A2T;
  unsigned short* const s_a3t = s_pool + P_A3T;
  float*          const s_a4  = (float*)(s_pool + P_A4);
  unsigned short* const s_in  = s_pool + P_SIN;

  const int tid   = threadIdx.x;
  const int wave  = tid >> 5;
  const int lane  = tid & 31;
  const int l15   = lane & 15;
  const int mh    = (lane & 16) ? 8 : 0;
  const int half8 = mh;
  const int trash = P_TRASH + lane;               // per-lane ushort trash slot
  float* const trashf = (float*)(s_pool + P_TRASHF) + lane;

  const int frame = blockIdx.x;
  const int traj  = blockIdx.y;
  const float* in  = (traj ? traj_j : traj_i) + (long)frame * 28224;  // [84][84][4]
  const float* g26 = (traj ? g26j : g26i) + (long)frame * 676;
  const float* g11 = (traj ? g11j : g11i) + (long)frame * 121;
  __builtin_prefetch(in, 0, 1);

  // ---- zero a1p (pad cols must read 0 in gaze26) + stage conv1 weights ----
  for (int idx = tid; idx < 13312 / 8; idx += 256)
    *(uint4*)(s_a1p + idx * 8) = uint4{0, 0, 0, 0};
  // W1F: K order k = (kh*8 + kw)*4 + ci, kw 0..7 (kw==7 zero-padded) -> K=224
  for (int idx = tid; idx < 7 * 512; idx += 256) {
    int i = idx & 15, ln = (idx >> 4) & 31, kt = idx >> 9;
    int k = kt * 32 + ((ln & 16) ? 8 : 0) + (i & 7) + ((i & 8) ? 16 : 0), n = ln & 15;
    int tap = k >> 2, ci = k & 3, kh = tap >> 3, kw = tap & 7;
    float v = (kw < 7) ? w1[((n * 4 + ci) * 7 + kh) * 7 + kw] : 0.f;
    s_pool[P_W1F + idx] = f2bf(v);
  }

  // ---- conv1 in two row-band passes: rows [0,46) tiles 0..21, rows [39,82) tiles 22..42 ----
#pragma unroll 1
  for (int pass = 0; pass < 2; ++pass) {
    const int r0  = pass ? 39 : 0;
    const int nr  = pass ? 43 : 46;
    const int mt0 = pass ? 22 : 0;
    const int mt1 = pass ? 43 : 22;
    __syncthreads();
    {  // stage input band: coalesced float4 -> packed bf16x4
      const float4* src = (const float4*)(in + r0 * 336);
      const int n4 = nr * 84;
      for (int idx = tid; idx < n4; idx += 256) {
        float4 v = src[idx];
        unsigned u0 = (unsigned)f2bf(v.x) | ((unsigned)f2bf(v.y) << 16);
        unsigned u1 = (unsigned)f2bf(v.z) | ((unsigned)f2bf(v.w) << 16);
        *(uint2*)(s_in + idx * 4) = make_uint2(u0, u1);
      }
    }
    __syncthreads();
    for (int mt = mt0 + wave; mt < mt1; mt += 8) {
      int m = mt * 16 + l15;
      int m_c = imin(m, 675);
      int oh = m_c / 26, ow = m_c % 26;
      const unsigned short* ab = s_in + ((oh * 3 - r0) * 84 + ow * 3) * 4 + half8;
      v8f acc = vbroad(b1[l15]);
#pragma unroll
      for (int kt = 0; kt < 7; ++kt) {   // kt == kh
        const unsigned short* p = ab + kt * 336;
        BF16x16 a;
        ld4(a, 0, p); ld4(a, 4, p + 4); ld4(a, 8, p + 16); ld4(a, 12, p + 20);
        acc = wmma_bf16(a, s_pool + P_W1F + (kt * 32 + lane) * 16, acc);
      }
#pragma unroll
      for (int v = 0; v < 8; ++v) {
        int md = mt * 16 + v + mh;
        int addr = (md < 676) ? (P_A1P + l15 * 832 + (md / 26) * 32 + (md % 26)) : trash;
        s_pool[addr] = f2bf(leaky(acc[v]));
      }
    }
  }
  __syncthreads();

  // ---- stage gaze26 fragments (overlay dead input band) ----
  for (int idx = tid; idx < 512; idx += 256) {
    int i = idx & 15, ln = idx >> 4;
    int k = ((ln & 16) ? 8 : 0) + (i & 7) + ((i & 8) ? 16 : 0), n = ln & 15;
    float v0 = 0.f, v1 = 0.f;
    if (k < 26) {
      v0 = (g26[k * 26 + n] - 0.3f) * (1.f / 0.21f);
      if (n + 16 < 26) v1 = (g26[k * 26 + n + 16] - 0.3f) * (1.f / 0.21f);
    }
    s_pool[P_K26F0 + idx] = f2bf(v0);
    s_pool[P_K26F1 + idx] = f2bf(v1);
  }
  __syncthreads();

  // ---- gaze26: A from a1p (contiguous, pad cols zero), D -> a1t [pos][ci] ----
  for (int mt = wave; mt < 26; mt += 8) {
    int m = mt * 16 + l15, c = m / 26, h = m % 26;
    const unsigned short* ab = s_a1p + (c * 26 + h) * 32 + half8;
    BF16x16 a; ld8(a, 0, ab); ld8(a, 8, ab + 16);
    v8f d0 = wmma_bf16(a, s_pool + P_K26F0 + lane * 16, vbroad(0.f));
    v8f d1 = wmma_bf16(a, s_pool + P_K26F1 + lane * 16, vbroad(0.f));
#pragma unroll
    for (int v = 0; v < 8; ++v) {
      int md = mt * 16 + v + mh, cc = md / 26, hh = md % 26;
      s_a1t[(hh * 26 + l15) * 16 + cc] = f2bf(d0[v]);
      int addr = (l15 < 10) ? (P_A1T + (hh * 26 + 16 + l15) * 16 + cc) : trash;
      s_pool[addr] = f2bf(d1[v]);
    }
  }
  __syncthreads();

  // ---- stage conv2/3/4 weights + K11 + zero a2p (overlay dead regions) ----
  for (int idx = tid; idx < 2816 / 8; idx += 256)
    *(uint4*)(s_a2p + idx * 8) = uint4{0, 0, 0, 0};
  for (int idx = tid; idx < 512; idx += 256) {
    int i = idx & 15, ln = idx >> 4;
    int k = ((ln & 16) ? 8 : 0) + (i & 7) + ((i & 8) ? 16 : 0), n = ln & 15;
    float v = (k < 11 && n < 11) ? (g11[k * 11 + n] - 0.3f) * (1.f / 0.21f) : 0.f;
    s_pool[P_K11F + idx] = f2bf(v);
  }
  // W2F: k = tap*16 + ci, 26 taps (tap 25 zero) -> K=416
  for (int idx = tid; idx < 13 * 512; idx += 256) {
    int i = idx & 15, ln = (idx >> 4) & 31, kt = idx >> 9;
    int k = kt * 32 + ((ln & 16) ? 8 : 0) + (i & 7) + ((i & 8) ? 16 : 0), n = ln & 15;
    int tap = k >> 4, ci = k & 15;
    float v = (tap < 25) ? w2[((n * 16 + ci) * 5 + tap / 5) * 5 + tap % 5] : 0.f;
    s_pool[P_W2F + idx] = f2bf(v);
  }
  // W3F/W4F: k = tap*16 + ci, 10 taps (tap 9 zero) -> K=160
  for (int idx = tid; idx < 5 * 512; idx += 256) {
    int i = idx & 15, ln = (idx >> 4) & 31, kt = idx >> 9;
    int k = kt * 32 + ((ln & 16) ? 8 : 0) + (i & 7) + ((i & 8) ? 16 : 0), n = ln & 15;
    int tap = k >> 4, ci = k & 15;
    float v3 = 0.f, v4 = 0.f;
    if (tap < 9) {
      int off = ((n * 16 + ci) * 3 + tap / 3) * 3 + tap % 3;
      v3 = w3[off]; v4 = w4[off];
    }
    s_pool[P_W3F + idx] = f2bf(v3);
    s_pool[P_W4F + idx] = f2bf(v4);
  }
  __syncthreads();

  // ---- conv2: A from a1t [pos][ci] (2x16B per frag), D -> a2p [c][h][16] ----
  for (int mt = wave; mt < 8; mt += 8) {
    int m = mt * 16 + l15;
    int m_c = imin(m, 120);
    int base = (m_c / 11) * 52 + (m_c % 11) * 2;   // (oh*2)*26 + ow*2
    const unsigned short* ab = s_a1t + base * 16 + half8;
    v8f acc = vbroad(b2[l15]);
#pragma unroll
    for (int kt = 0; kt < 13; ++kt) {
      BF16x16 a;
      ld8(a, 0, ab + dpos2(2 * kt) * 16);
      ld8(a, 8, ab + dpos2(2 * kt + 1) * 16);
      acc = wmma_bf16(a, s_pool + P_W2F + (kt * 32 + lane) * 16, acc);
    }
#pragma unroll
    for (int v = 0; v < 8; ++v) {
      int md = mt * 16 + v + mh;
      int addr = (md < 121) ? (P_A2P + l15 * 176 + (md / 11) * 16 + (md % 11)) : trash;
      s_pool[addr] = f2bf(leaky(acc[v]));
    }
  }
  __syncthreads();

  // ---- gaze11: A from a2p (contiguous, pad zero; k>=16 -> 0), D -> a2t [pos][ci] ----
  for (int mt = wave; mt < 11; mt += 8) {
    int m = mt * 16 + l15, c = m / 11, h = m % 11;
    const unsigned short* ab = s_a2p + (c * 11 + h) * 16 + half8;
    BF16x16 a; ld8(a, 0, ab);
#pragma unroll
    for (int i = 8; i < 16; ++i) a.h[i] = 0;
    v8f d0 = wmma_bf16(a, s_pool + P_K11F + lane * 16, vbroad(0.f));
#pragma unroll
    for (int v = 0; v < 8; ++v) {
      int md = mt * 16 + v + mh, cc = md / 11, hh = md % 11;
      int addr = (l15 < 11) ? (P_A2T + (hh * 11 + l15) * 16 + cc) : trash;
      s_pool[addr] = f2bf(d0[v]);
    }
  }
  __syncthreads();

  // ---- conv3: A from a2t, D -> a3t [pos][ci] ----
  for (int mt = wave; mt < 6; mt += 8) {
    int m = mt * 16 + l15;
    int m_c = imin(m, 80);
    int base = (m_c / 9) * 11 + (m_c % 9);
    const unsigned short* ab = s_a2t + base * 16 + half8;
    v8f acc = vbroad(b3[l15]);
#pragma unroll
    for (int kt = 0; kt < 5; ++kt) {
      BF16x16 a;
      ld8(a, 0, ab + dpos3(2 * kt) * 16);
      ld8(a, 8, ab + dpos3(2 * kt + 1) * 16);
      acc = wmma_bf16(a, s_pool + P_W3F + (kt * 32 + lane) * 16, acc);
    }
#pragma unroll
    for (int v = 0; v < 8; ++v) {
      int md = mt * 16 + v + mh;
      int addr = (md < 81) ? (P_A3T + md * 16 + l15) : trash;
      s_pool[addr] = f2bf(leaky(acc[v]));
    }
  }
  __syncthreads();

  // ---- conv4: A from a3t, D -> a4 f32 [c][49] ----
  for (int mt = wave; mt < 4; mt += 8) {
    int m = mt * 16 + l15;
    int m_c = imin(m, 48);
    int base = (m_c / 7) * 9 + (m_c % 7);
    const unsigned short* ab = s_a3t + base * 16 + half8;
    v8f acc = vbroad(b4[l15]);
#pragma unroll
    for (int kt = 0; kt < 5; ++kt) {
      BF16x16 a;
      ld8(a, 0, ab + dpos4(2 * kt) * 16);
      ld8(a, 8, ab + dpos4(2 * kt + 1) * 16);
      acc = wmma_bf16(a, s_pool + P_W4F + (kt * 32 + lane) * 16, acc);
    }
#pragma unroll
    for (int v = 0; v < 8; ++v) {
      int md = mt * 16 + v + mh;
      float* dst = (md < 49) ? (s_a4 + l15 * 49 + md) : trashf;
      *dst = leaky(acc[v]);
    }
  }
  __syncthreads();

  // ---- epilogue: xf bf16 (K padded to 800) + per-frame channel-sum cm ----
  unsigned short* xf = ws_xf + ((long)traj * 2048 + frame) * 800;
  for (int i = tid; i < 800; i += 256) xf[i] = (i < 784) ? f2bf(s_a4[i]) : (unsigned short)0;
  float* cm = ws_cm + ((long)traj * 2048 + frame) * 49;
  for (int s = tid; s < 49; s += 256) {
    float a = 0.f;
#pragma unroll
    for (int c = 0; c < 16; ++c) a += s_a4[c * 49 + s];
    cm[s] = a;
  }
}

// ---------------------------------------------------------------------------
// fc_prep: fc1_w (64x784 f32) -> fragment-ordered bf16 [nt][kt][lane][16]
// ---------------------------------------------------------------------------
__global__ __launch_bounds__(256)
void fc_prep(const float* __restrict__ fc1w, unsigned short* __restrict__ wfrag) {
  int idx = blockIdx.x * 256 + threadIdx.x;   // 200*256 = 51200 exact
  int i = idx & 15, t = idx >> 4;
  int lane = t & 31; t >>= 5;
  int kt = t % 25, nt = t / 25;
  int k = kt * 32 + ((lane & 16) ? 8 : 0) + (i & 7) + ((i & 8) ? 16 : 0);
  int n = nt * 16 + (lane & 15);
  float v = (k < 784) ? fc1w[n * 784 + k] : 0.f;
  wfrag[idx] = f2bf(v);
}

// ---------------------------------------------------------------------------
// fc_kernel: FC1 4096x784x64 bf16 WMMA + leaky + FC2 dot -> r[4096]
// ---------------------------------------------------------------------------
__global__ __launch_bounds__(128)
void fc_kernel(const unsigned short* __restrict__ xf,
               const unsigned short* __restrict__ wfrag,
               const float* __restrict__ fc1b,
               const float* __restrict__ fc2w, const float* __restrict__ fc2b,
               float* __restrict__ r) {
  __shared__ float s_h[16 * 64];
  const int tid   = threadIdx.x;
  const int wave  = tid >> 5;
  const int lane  = tid & 31;
  const int l15   = lane & 15;
  const int mh    = (lane & 16) ? 8 : 0;
  const int half8 = mh;
  const int mt    = blockIdx.x;
  const int n0    = wave * 16;

  const unsigned short* arow = xf + (long)(mt * 16 + l15) * 800;
  const unsigned short* brow = wfrag + (long)wave * 25 * 512;

  v8f acc = vbroad(fc1b[n0 + l15]);
#pragma unroll
  for (int kt = 0; kt < 25; ++kt) {
    BF16x16 a;
    ld8(a, 0, arow + kt * 32 + half8);
    ld8(a, 8, arow + kt * 32 + half8 + 16);
    acc = wmma_bf16(a, brow + (kt * 32 + lane) * 16, acc);
  }
#pragma unroll
  for (int v = 0; v < 8; ++v) s_h[(v + mh) * 64 + n0 + l15] = leaky(acc[v]);
  __syncthreads();

  if (tid < 16) {
    float s = fc2b[0];
#pragma unroll
    for (int n = 0; n < 64; ++n) s += s_h[tid * 64 + n] * fc2w[n];
    r[mt * 16 + tid] = s;
  }
}

// ---------------------------------------------------------------------------
__global__ __launch_bounds__(64)
void rb_reduce(const float* __restrict__ r, float* __restrict__ out) {
  __shared__ float s0[64], s1[64], s2[64];
  const int b = blockIdx.x, t = threadIdx.x;
  float ri = r[t * 32 + b];
  float rj = r[2048 + t * 32 + b];
  s0[t] = ri; s1[t] = rj; s2[t] = fabsf(ri) + fabsf(rj);
  __syncthreads();
  for (int s = 32; s > 0; s >>= 1) {
    if (t < s) { s0[t] += s0[t + s]; s1[t] += s1[t + s]; s2[t] += s2[t + s]; }
    __syncthreads();
  }
  if (t == 0) { out[b] = s0[0]; out[32 + b] = s1[0]; out[64 + b] = s2[0]; }
}

__global__ __launch_bounds__(256)
void cm_norm(const float* __restrict__ cm, float* __restrict__ out) {
  __shared__ float smn[256], smx[256];
  const int t = blockIdx.x, traj = blockIdx.y, tid = threadIdx.x;
  const float* src = cm + ((long)traj * 2048 + (long)t * 32) * 49;
  float mn = 3.4e38f, mx = -3.4e38f;
  for (int i = tid; i < 1568; i += 256) {
    float v = src[i]; mn = fminf(mn, v); mx = fmaxf(mx, v);
  }
  smn[tid] = mn; smx[tid] = mx; __syncthreads();
  for (int s = 128; s > 0; s >>= 1) {
    if (tid < s) {
      smn[tid] = fminf(smn[tid], smn[tid + s]);
      smx[tid] = fmaxf(smx[tid], smx[tid + s]);
    }
    __syncthreads();
  }
  mn = smn[0]; mx = smx[0];
  const float inv = 1.f / (mx - mn);
  float* dst = out + 96 + (long)traj * 100352 + (long)t * 1568;
  for (int i = tid; i < 1568; i += 256) dst[i] = (src[i] - mn) * inv;
}

// ---------------------------------------------------------------------------
extern "C" void kernel_launch(void* const* d_in, const int* in_sizes, int n_in,
                              void* d_out, int out_size, void* d_ws, size_t ws_size,
                              hipStream_t stream) {
  const float* traj_i = (const float*)d_in[0];
  const float* traj_j = (const float*)d_in[1];
  const float* g26i   = (const float*)d_in[2];
  const float* g11i   = (const float*)d_in[3];
  const float* g26j   = (const float*)d_in[4];
  const float* g11j   = (const float*)d_in[5];
  const float* w1 = (const float*)d_in[6];  const float* b1 = (const float*)d_in[7];
  const float* w2 = (const float*)d_in[8];  const float* b2 = (const float*)d_in[9];
  const float* w3 = (const float*)d_in[10]; const float* b3 = (const float*)d_in[11];
  const float* w4 = (const float*)d_in[12]; const float* b4 = (const float*)d_in[13];
  const float* fc1w = (const float*)d_in[14]; const float* fc1b = (const float*)d_in[15];
  const float* fc2w = (const float*)d_in[16]; const float* fc2b = (const float*)d_in[17];

  // workspace: xf bf16 [4096][800] | cm f32 [4096][49] | r f32 [4096] | fc1w frags
  const size_t XF_BYTES = (size_t)4096 * 800 * 2;
  const size_t CM_BYTES = (size_t)4096 * 49 * 4;
  const size_t R_BYTES  = (size_t)4096 * 4;
  unsigned short* ws_xf = (unsigned short*)d_ws;
  float* ws_cm = (float*)((char*)d_ws + XF_BYTES);
  float* ws_r  = (float*)((char*)d_ws + XF_BYTES + CM_BYTES);
  unsigned short* ws_wf = (unsigned short*)((char*)d_ws + XF_BYTES + CM_BYTES + R_BYTES);

  fc_prep<<<200, 256, 0, stream>>>(fc1w, ws_wf);
  conv_frames<<<dim3(2048, 2), 256, 0, stream>>>(
      traj_i, traj_j, g26i, g11i, g26j, g11j,
      w1, b1, w2, b2, w3, b3, w4, b4, ws_xf, ws_cm);
  fc_kernel<<<256, 128, 0, stream>>>(ws_xf, ws_wf, fc1b, fc2w, fc2b, ws_r);
  rb_reduce<<<32, 64, 0, stream>>>(ws_r, (float*)d_out);
  cm_norm<<<dim3(64, 2), 256, 0, stream>>>(ws_cm, (float*)d_out);

  (void)in_sizes; (void)n_in; (void)out_size; (void)ws_size;
}
